// NeuralODELM_38817914421762
// MI455X (gfx1250) — matrix operand order repaired and checked by reading, not compiled, not run on hardware
//
#include <hip/hip_runtime.h>
#include <hip/hip_bf16.h>
#include <math.h>

// ---------------------------------------------------------------------------
// NeuralODE LM for gfx1250 (MI455X): bf16 WMMA GEMMs + f32 state RK4 + tied head
// Round 4: AS(1) fragment loads via clang ext_vector types (HIP vector classes
//          cannot bind AS(1) references); K-chunked loop, immediate offsets
// ---------------------------------------------------------------------------

typedef __bf16 v16bf __attribute__((ext_vector_type(16)));
typedef float  v8f   __attribute__((ext_vector_type(8)));
typedef unsigned int u32x4 __attribute__((ext_vector_type(4)));
typedef float        f32x4 __attribute__((ext_vector_type(4)));

typedef __attribute__((address_space(1))) const u32x4 gu32x4;
typedef __attribute__((address_space(1))) const f32x4 gf32x4;

#define HID    512
#define FF     2048
#define ROWS   2048           // B*S = 2*1024
#define VOCABN 50257
#define NSTEPS 8
#define KCHUNK 256            // K per outer iter; 8 unrolled micro-steps of 32

// -------------------------- embed gather -----------------------------------
__global__ void gather_kernel(const int* __restrict__ tokens,
                              const float* __restrict__ embed,
                              float* __restrict__ h) {
    int row = blockIdx.x;              // 0..ROWS-1
    int t   = threadIdx.x;             // 0..127, each thread moves a float4
    int tok = tokens[row];
    const float4* src = reinterpret_cast<const float4*>(embed + (size_t)tok * HID);
    float4* dst       = reinterpret_cast<float4*>(h + (size_t)row * HID);
    dst[t] = src[t];
}

// ---------------- weight convert + transpose (f32 [K,N] -> bf16 [N,K]) -----
__global__ void cvt_transpose_kernel(const float* __restrict__ w,
                                     __bf16* __restrict__ wt,
                                     int K, int N) {
    int idx = blockIdx.x * blockDim.x + threadIdx.x;  // output index: n*K + k
    if (idx >= K * N) return;
    int n = idx / K;
    int k = idx - n * K;
    wt[idx] = (__bf16)w[(size_t)k * N + n];
}

// ------------------- row LayerNorm -> bf16 (one wave per row) --------------
__global__ void ln_bf16_kernel(const float* __restrict__ x,
                               const float* __restrict__ g,
                               const float* __restrict__ b,
                               __bf16* __restrict__ out, int rows) {
    int wave = (blockIdx.x * blockDim.x + threadIdx.x) >> 5;
    int lane = threadIdx.x & 31;
    if (wave >= rows) return;
    const float* xr = x + (size_t)wave * HID;
    float4 v[4];
    float s = 0.f, ss = 0.f;
#pragma unroll
    for (int i = 0; i < 4; ++i) {
        v[i] = reinterpret_cast<const float4*>(xr)[lane * 4 + i];
        s  += v[i].x + v[i].y + v[i].z + v[i].w;
        ss += v[i].x * v[i].x + v[i].y * v[i].y + v[i].z * v[i].z + v[i].w * v[i].w;
    }
#pragma unroll
    for (int off = 16; off > 0; off >>= 1) {
        s  += __shfl_xor(s,  off, 32);
        ss += __shfl_xor(ss, off, 32);
    }
    float mean = s * (1.0f / HID);
    float var  = ss * (1.0f / HID) - mean * mean;
    float rstd = rsqrtf(var + 1e-5f);
    __bf16* orow = out + (size_t)wave * HID;
#pragma unroll
    for (int i = 0; i < 4; ++i) {
        float e[4] = {v[i].x, v[i].y, v[i].z, v[i].w};
#pragma unroll
        for (int c = 0; c < 4; ++c) {
            int idx = lane * 16 + i * 4 + c;
            float o = (e[c] - mean) * rstd * g[idx] + b[idx];
            orow[idx] = (__bf16)o;
        }
    }
}

// ---------------------------- WMMA GEMM ------------------------------------
// C[M,N] = A[M,K](bf16) * Bt[N,K]  (Bt row-major N x K, i.e. B pre-transposed)
// Wave tile 32x64 (2x4 wmma accs). Block = 8 waves as 2(M) x 4(N) -> 64x256.
// EPI 0: C += bias[n]; GELU(exact); store bf16
// EPI 1: C += bias[n]; store f32
// EPI 2: head: Bt is f32 (embed_w [V,H]), cvt in-register; N-masked f32 store
__device__ __forceinline__ float gelu_exact(float v) {
    return 0.5f * v * (1.0f + erff(v * 0.70710678118654752f));
}

template <int EPI>
__global__ __launch_bounds__(256, 1)
void wmma_gemm_kernel(const __bf16* __restrict__ A,
                      const void* __restrict__ Bv,
                      void* __restrict__ Cv,
                      const float* __restrict__ bias,
                      int M, int N, int K, int lda, int ldb, int ldc) {
    const int lane = threadIdx.x & 31;
    const int wave = threadIdx.x >> 5;
    const int m0 = blockIdx.y * 64  + (wave >> 2) * 32;   // 2 waves in M
    const int n0 = blockIdx.x * 256 + (wave & 3) * 64;    // 4 waves in N

    const int ar = lane & 15;      // row/col within 16
    const int ah = lane >> 4;      // half-wave selector

    // ---- fixed per-lane fragment base pointers (global address space) ------
    // A: lane<16 holds K+0..15; lane>=16 holds K+16..31
    gu32x4* ap0 = (gu32x4*)(A + (size_t)(m0 + 0  + ar) * lda + ah * 16);
    gu32x4* ap1 = (gu32x4*)(A + (size_t)(m0 + 16 + ar) * lda + ah * 16);

    // B: lane<16 holds K+0..15; lane>=16 holds K+16..31
    gu32x4* bp[4];    // bf16 path
    gf32x4* bpf[4];   // f32 path (head)
#pragma unroll
    for (int ni = 0; ni < 4; ++ni) {
        if constexpr (EPI == 2) {
            int row = n0 + ni * 16 + ar;
            row = (row < N) ? row : (N - 1);
            bpf[ni] = (gf32x4*)(reinterpret_cast<const float*>(Bv) +
                                (size_t)row * ldb + ah * 16);
        } else {
            bp[ni] = (gu32x4*)(reinterpret_cast<const __bf16*>(Bv) +
                               (size_t)(n0 + ni * 16 + ar) * ldb + ah * 16);
        }
    }

    v8f acc[2][4] = {};

    for (int kc = 0; kc < K; kc += KCHUNK) {
#pragma unroll
        for (int kk = 0; kk < KCHUNK; kk += 32) {
            const int qi = kk >> 3;          // u32x4 index (8 bf16 per u32x4)
            v16bf a0, a1, bfr[4];
            {
                union { v16bf v; u32x4 q[2]; } u;
                u.q[0] = ap0[qi];            // K = kc+kk + ah*16 + 0..7
                u.q[1] = ap0[qi + 1];        // K = kc+kk + ah*16 + 8..15
                a0 = u.v;
            }
            {
                union { v16bf v; u32x4 q[2]; } u;
                u.q[0] = ap1[qi];
                u.q[1] = ap1[qi + 1];
                a1 = u.v;
            }
#pragma unroll
            for (int ni = 0; ni < 4; ++ni) {
                if constexpr (EPI == 2) {
                    const int fi = kk >> 2;  // f32x4 index (4 f32 per f32x4)
                    f32x4 ff[4];
                    ff[0] = bpf[ni][fi + 0];
                    ff[1] = bpf[ni][fi + 1];
                    ff[2] = bpf[ni][fi + 2];
                    ff[3] = bpf[ni][fi + 3];
                    v16bf bv;
#pragma unroll
                    for (int e = 0; e < 16; ++e)
                        bv[e] = (__bf16)ff[e >> 2][e & 3];
                    bfr[ni] = bv;
                } else {
                    union { v16bf v; u32x4 q[2]; } u;
                    u.q[0] = bp[ni][qi];
                    u.q[1] = bp[ni][qi + 1];
                    bfr[ni] = u.v;
                }
            }
#pragma unroll
            for (int ni = 0; ni < 4; ++ni) {
                acc[0][ni] = __builtin_amdgcn_wmma_f32_16x16x32_bf16(
                    false, a0, false, bfr[ni], (short)0, acc[0][ni], false, false);
                acc[1][ni] = __builtin_amdgcn_wmma_f32_16x16x32_bf16(
                    false, a1, false, bfr[ni], (short)0, acc[1][ni], false, false);
            }
        }
        // advance bases once per chunk: KCHUNK bf16 = KCHUNK/8 u32x4
        ap0 += KCHUNK / 8;
        ap1 += KCHUNK / 8;
#pragma unroll
        for (int ni = 0; ni < 4; ++ni) {
            if constexpr (EPI == 2) bpf[ni] += KCHUNK / 4;
            else                    bp[ni]  += KCHUNK / 8;
        }
    }

    // ---- epilogue: VGPR r -> M = r + (lane>=16 ? 8 : 0), N = lane&15 -------
    const int rhi = ah * 8;
#pragma unroll
    for (int mi = 0; mi < 2; ++mi) {
#pragma unroll
        for (int ni = 0; ni < 4; ++ni) {
#pragma unroll
            for (int r = 0; r < 8; ++r) {
                int m = m0 + mi * 16 + r + rhi;
                int n = n0 + ni * 16 + ar;
                float v = acc[mi][ni][r];
                if constexpr (EPI == 0) {
                    v += bias[n];
                    v = gelu_exact(v);
                    reinterpret_cast<__bf16*>(Cv)[(size_t)m * ldc + n] = (__bf16)v;
                } else if constexpr (EPI == 1) {
                    v += bias[n];
                    reinterpret_cast<float*>(Cv)[(size_t)m * ldc + n] = v;
                } else {
                    if (n < N)
                        reinterpret_cast<float*>(Cv)[(size_t)m * ldc + n] = v;
                }
            }
        }
    }
}

// ------------------------- RK4 combine kernels -----------------------------
__global__ void axpy_kernel(float* __restrict__ out, const float* __restrict__ x,
                            const float* __restrict__ y, float a, int n4) {
    int i = blockIdx.x * blockDim.x + threadIdx.x;
    if (i >= n4) return;
    float4 xv = reinterpret_cast<const float4*>(x)[i];
    float4 yv = reinterpret_cast<const float4*>(y)[i];
    float4 o;
    o.x = xv.x + a * yv.x; o.y = xv.y + a * yv.y;
    o.z = xv.z + a * yv.z; o.w = xv.w + a * yv.w;
    reinterpret_cast<float4*>(out)[i] = o;
}

__global__ void acc_kernel(float* __restrict__ acc, const float* __restrict__ k,
                           float c, int init, int n4) {
    int i = blockIdx.x * blockDim.x + threadIdx.x;
    if (i >= n4) return;
    float4 kv = reinterpret_cast<const float4*>(k)[i];
    float4 av;
    if (init) { av.x = av.y = av.z = av.w = 0.f; }
    else      { av = reinterpret_cast<const float4*>(acc)[i]; }
    av.x += c * kv.x; av.y += c * kv.y; av.z += c * kv.z; av.w += c * kv.w;
    reinterpret_cast<float4*>(acc)[i] = av;
}

// ------------------------------- host --------------------------------------
extern "C" void kernel_launch(void* const* d_in, const int* in_sizes, int n_in,
                              void* d_out, int out_size, void* d_ws, size_t ws_size,
                              hipStream_t stream) {
    const int*   tokens = (const int*)d_in[0];
    const float* embed  = (const float*)d_in[1];
    const float* ln1g   = (const float*)d_in[2];
    const float* ln1b   = (const float*)d_in[3];
    const float* w1     = (const float*)d_in[4];
    const float* b1     = (const float*)d_in[5];
    const float* w2     = (const float*)d_in[6];
    const float* b2     = (const float*)d_in[7];
    const float* nfg    = (const float*)d_in[8];
    const float* nfb    = (const float*)d_in[9];
    float* out = (float*)d_out;

    char* ws = (char*)d_ws;
    const size_t MB = 1024 * 1024;
    float*  h      = (float*)(ws + 0 * MB);    // 4 MB  [ROWS,HID] f32
    float*  ytmp   = (float*)(ws + 4 * MB);    // 4 MB
    float*  kbuf   = (float*)(ws + 8 * MB);    // 4 MB
    float*  accb   = (float*)(ws + 12 * MB);   // 4 MB
    __bf16* yln    = (__bf16*)(ws + 16 * MB);  // 2 MB  [ROWS,HID] bf16
    __bf16* hidden = (__bf16*)(ws + 18 * MB);  // 8 MB  [ROWS,FF]  bf16
    __bf16* w1t    = (__bf16*)(ws + 26 * MB);  // 2 MB  [FF,HID]   bf16
    __bf16* w2t    = (__bf16*)(ws + 28 * MB);  // 2 MB  [HID,FF]   bf16
    __bf16* hln    = (__bf16*)(ws + 30 * MB);  // 2 MB  [ROWS,HID] bf16

    const int n4 = ROWS * HID / 4;
    const int nblk = (n4 + 255) / 256;

    // embed gather + weight convert/transpose (weights constant per launch)
    gather_kernel<<<ROWS, 128, 0, stream>>>(tokens, embed, h);
    cvt_transpose_kernel<<<(HID * FF + 255) / 256, 256, 0, stream>>>(w1, w1t, HID, FF);
    cvt_transpose_kernel<<<(FF * HID + 255) / 256, 256, 0, stream>>>(w2, w2t, FF, HID);

    // ode_func: kout = W2 * GELU(W1 * LN(y) + b1) + b2
    auto eval_f = [&](const float* y, float* kout) {
        ln_bf16_kernel<<<ROWS / 8, 256, 0, stream>>>(y, ln1g, ln1b, yln, ROWS);
        dim3 g1(FF / 256, ROWS / 64);         // N=2048, M=2048
        wmma_gemm_kernel<0><<<g1, 256, 0, stream>>>(yln, w1t, hidden, b1,
                                                    ROWS, FF, HID, HID, HID, FF);
        dim3 g2(HID / 256, ROWS / 64);        // N=512, M=2048
        wmma_gemm_kernel<1><<<g2, 256, 0, stream>>>(hidden, w2t, kout, b2,
                                                    ROWS, HID, FF, FF, FF, HID);
    };

    // fixed-step RK4, NSTEPS steps over [0,1]
    const float dt = 1.0f / NSTEPS;
    for (int s = 0; s < NSTEPS; ++s) {
        eval_f(h, kbuf);                                                    // k1
        acc_kernel<<<nblk, 256, 0, stream>>>(accb, kbuf, 1.0f, 1, n4);
        axpy_kernel<<<nblk, 256, 0, stream>>>(ytmp, h, kbuf, 0.5f * dt, n4);
        eval_f(ytmp, kbuf);                                                 // k2
        acc_kernel<<<nblk, 256, 0, stream>>>(accb, kbuf, 2.0f, 0, n4);
        axpy_kernel<<<nblk, 256, 0, stream>>>(ytmp, h, kbuf, 0.5f * dt, n4);
        eval_f(ytmp, kbuf);                                                 // k3
        acc_kernel<<<nblk, 256, 0, stream>>>(accb, kbuf, 2.0f, 0, n4);
        axpy_kernel<<<nblk, 256, 0, stream>>>(ytmp, h, kbuf, dt, n4);
        eval_f(ytmp, kbuf);                                                 // k4
        acc_kernel<<<nblk, 256, 0, stream>>>(accb, kbuf, 1.0f, 0, n4);
        axpy_kernel<<<nblk, 256, 0, stream>>>(h, h, accb, dt / 6.0f, n4);
    }

    // final LN + tied head: logits = LN(h) @ embed^T
    ln_bf16_kernel<<<ROWS / 8, 256, 0, stream>>>(h, nfg, nfb, hln, ROWS);
    dim3 gh((VOCABN + 255) / 256, ROWS / 64);
    wmma_gemm_kernel<2><<<gh, 256, 0, stream>>>(hln, embed, out, nullptr,
                                                ROWS, VOCABN, HID, HID, HID, VOCABN);
}
